// PartialEncoder_84232898609398
// MI455X (gfx1250) — matrix operand order repaired and checked by reading, not compiled, not run on hardware
//
#include <hip/hip_runtime.h>

typedef __bf16 bf16;
typedef bf16  v16bf __attribute__((ext_vector_type(16)));
typedef bf16  v8bf  __attribute__((ext_vector_type(8)));
typedef float v8f   __attribute__((ext_vector_type(8)));

#define DEV __device__ __forceinline__

enum { ACT_NONE = 0, ACT_LRELU = 1, ACT_RELU = 2, ACT_GELU = 3, ACT_SIGMOID = 4 };

DEV unsigned short f2bf_bits(float f) {
  union { float f; unsigned u; } c; c.f = f;
  unsigned r = c.u + 0x7fffu + ((c.u >> 16) & 1u);   // round-to-nearest-even
  return (unsigned short)(r >> 16);
}

// pack two floats into one dword of bf16 (lo, hi). Prefer the HW packed convert.
DEV unsigned pack2bf(float lo, float hi) {
#if __has_builtin(__builtin_amdgcn_cvt_pk_bf16_f32)
  auto r = __builtin_amdgcn_cvt_pk_bf16_f32(lo, hi);   // v_cvt_pk_bf16_f32
  return __builtin_bit_cast(unsigned, r);
#else
  return ((unsigned)f2bf_bits(hi) << 16) | f2bf_bits(lo);
#endif
}

DEV float lrelu_f(float v) { return v >= 0.f ? v : 0.2f * v; }

// ---------------------------------------------------------------------------
// Generic batched GEMM: C = act((alpha * A@B + bias)[*sc2+sh2]) + resid
// A: (M,K) f32, B: (K,N) f32 (or transB: B[k][n]=Bsrc[n*ldb+k]).
// bf16 WMMA 16x16x32, block tile 64x128, 8 waves, 2x2 fragments per wave.
// REQUIRES: M % 64 == 0, K % 32 == 0, N % 8 == 0 (host guarantees).
// Vectorized b128 global loads + LDS double buffering (1 barrier per K-step).
// Two-level batching: z -> (outer = z/innerB, inner = z%innerB).
// ---------------------------------------------------------------------------
__global__ __launch_bounds__(256) void wmma_gemm(
    const float* __restrict__ A, long long lda, long long soA, long long siA,
    const float* __restrict__ B, long long ldb, long long soB, long long siB, int transB,
    float* __restrict__ C, long long ldc, long long soC, long long siC,
    int M, int N, int K, int innerB, float alpha,
    const float* __restrict__ bias, const float* __restrict__ sc2, const float* __restrict__ sh2,
    const float* __restrict__ resid, long long ldr, int act)
{
  constexpr int BM = 64, BN = 128, BK = 32;
  __shared__ alignas(64) bf16 As[2][BM * BK];
  __shared__ alignas(64) bf16 Bs[2][BN * BK];   // stored [n][k]

  int bz = blockIdx.z;
  long long bo = bz / innerB, bi = bz % innerB;
  const float* Ab = A + bo * soA + bi * siA;
  const float* Bb = B + bo * soB + bi * siB;
  float*       Cb = C + bo * soC + bi * siC;
  const float* Rb = resid ? (resid + bo * soC + bi * siC) : nullptr;

  int m0 = blockIdx.y * BM, n0 = blockIdx.x * BN;
  int tid = threadIdx.x, lane = tid & 31, wid = tid >> 5;
  int wm = wid & 1, wn = wid >> 1;        // 2 x 4 wave grid
  int hf = lane >> 4, lr = lane & 15;     // half-wave, lane row

  v8f acc[2][2];
  for (int mi = 0; mi < 2; ++mi)
    for (int ni = 0; ni < 2; ++ni) {
      v8f z;
      #pragma unroll
      for (int r = 0; r < 8; ++r) z[r] = 0.f;
      acc[mi][ni] = z;
    }

  const int ar  = tid >> 2;            // A row 0..63
  const int ac  = (tid & 3) * 8;       // A col segment
  const int bk2 = (tid >> 4) * 2;      // B (no-trans): k pair 0..30
  const int bn8 = (tid & 15) * 8;      //               n segment 0..120
  const int bn  = tid >> 1;            // B (trans): n 0..127
  const int bk16 = (tid & 1) * 16;     //            k segment 0 / 16

  float areg[8], breg[16];

  auto loadA = [&](int kb) {
    const float4* p = reinterpret_cast<const float4*>(Ab + (long long)(m0 + ar) * lda + kb + ac);
    float4 u = p[0], v = p[1];
    areg[0] = u.x; areg[1] = u.y; areg[2] = u.z; areg[3] = u.w;
    areg[4] = v.x; areg[5] = v.y; areg[6] = v.z; areg[7] = v.w;
  };
  auto loadB = [&](int kb) {
    if (!transB) {
      int gn = n0 + bn8;
      if (gn + 8 <= N) {
        const float* r0 = Bb + (long long)(kb + bk2) * ldb + gn;
        const float* r1 = r0 + ldb;
        float4 u0 = reinterpret_cast<const float4*>(r0)[0];
        float4 u1 = reinterpret_cast<const float4*>(r0)[1];
        float4 v0 = reinterpret_cast<const float4*>(r1)[0];
        float4 v1 = reinterpret_cast<const float4*>(r1)[1];
        breg[0] = u0.x; breg[1] = u0.y; breg[2] = u0.z; breg[3] = u0.w;
        breg[4] = u1.x; breg[5] = u1.y; breg[6] = u1.z; breg[7] = u1.w;
        breg[8]  = v0.x; breg[9]  = v0.y; breg[10] = v0.z; breg[11] = v0.w;
        breg[12] = v1.x; breg[13] = v1.y; breg[14] = v1.z; breg[15] = v1.w;
      } else {
        #pragma unroll
        for (int j = 0; j < 16; ++j) breg[j] = 0.f;
      }
    } else {
      int gn = n0 + bn;
      if (gn < N) {
        const float4* p = reinterpret_cast<const float4*>(Bb + (long long)gn * ldb + kb + bk16);
        float4 q0 = p[0], q1 = p[1], q2 = p[2], q3 = p[3];
        breg[0]  = q0.x; breg[1]  = q0.y; breg[2]  = q0.z; breg[3]  = q0.w;
        breg[4]  = q1.x; breg[5]  = q1.y; breg[6]  = q1.z; breg[7]  = q1.w;
        breg[8]  = q2.x; breg[9]  = q2.y; breg[10] = q2.z; breg[11] = q2.w;
        breg[12] = q3.x; breg[13] = q3.y; breg[14] = q3.z; breg[15] = q3.w;
      } else {
        #pragma unroll
        for (int j = 0; j < 16; ++j) breg[j] = 0.f;
      }
    }
  };
  auto storeTiles = [&](int buf) {
    // A: 8 floats -> 4 packed dwords -> one 16B LDS store
    uint4 ua;
    ua.x = pack2bf(areg[0], areg[1]);
    ua.y = pack2bf(areg[2], areg[3]);
    ua.z = pack2bf(areg[4], areg[5]);
    ua.w = pack2bf(areg[6], areg[7]);
    *reinterpret_cast<uint4*>(&As[buf][ar * BK + ac]) = ua;
    if (!transB) {
      // pair across k (k2 low, k2+1 high), scatter along n
      #pragma unroll
      for (int j = 0; j < 8; ++j)
        *reinterpret_cast<unsigned*>(&Bs[buf][(bn8 + j) * BK + bk2]) = pack2bf(breg[j], breg[8 + j]);
    } else {
      uint4 u0, u1;
      u0.x = pack2bf(breg[0],  breg[1]);  u0.y = pack2bf(breg[2],  breg[3]);
      u0.z = pack2bf(breg[4],  breg[5]);  u0.w = pack2bf(breg[6],  breg[7]);
      u1.x = pack2bf(breg[8],  breg[9]);  u1.y = pack2bf(breg[10], breg[11]);
      u1.z = pack2bf(breg[12], breg[13]); u1.w = pack2bf(breg[14], breg[15]);
      *reinterpret_cast<uint4*>(&Bs[buf][bn * BK + bk16])     = u0;
      *reinterpret_cast<uint4*>(&Bs[buf][bn * BK + bk16 + 8]) = u1;
    }
  };
  auto mma_step = [&](int buf) {
    v16bf afr[2], bfr[2];
    #pragma unroll
    for (int mi = 0; mi < 2; ++mi) {
      int row = wm * 32 + mi * 16 + lr;
      // A k order used here (same as B): lane holds K = i + 16*half for its row
      afr[mi] = *(const v16bf*)&As[buf][row * BK + hf * 16];
    }
    #pragma unroll
    for (int ni = 0; ni < 2; ++ni) {
      int nr = wn * 32 + ni * 16 + lr;
      // B k order used here (same as A): lane holds K = i + 16*half for its column
      bfr[ni] = *(const v16bf*)&Bs[buf][nr * BK + hf * 16];
    }
    #pragma unroll
    for (int mi = 0; mi < 2; ++mi)
      #pragma unroll
      for (int ni = 0; ni < 2; ++ni)
        acc[mi][ni] = __builtin_amdgcn_wmma_f32_16x16x32_bf16(
            false, afr[mi], false, bfr[ni], (short)0, acc[mi][ni], false, false);
  };

  loadA(0); loadB(0);
  storeTiles(0);
  __syncthreads();
  int cur = 0;
  for (int kb = 0; kb < K; kb += BK) {
    bool more = (kb + BK) < K;
    if (more) { loadA(kb + BK); loadB(kb + BK); }
    mma_step(cur);
    if (more) {
      storeTiles(cur ^ 1);
      __syncthreads();
      cur ^= 1;
    }
  }

  // C/D layout: VGPR r -> m = r + 8*half, n = lane&15
  #pragma unroll
  for (int mi = 0; mi < 2; ++mi) {
    #pragma unroll
    for (int ni = 0; ni < 2; ++ni) {
      int gn = n0 + wn * 32 + ni * 16 + lr;
      if (gn < N) {
        #pragma unroll
        for (int r = 0; r < 8; ++r) {
          int gm = m0 + wm * 32 + mi * 16 + r + 8 * hf;
          float v = alpha * acc[mi][ni][r];
          if (bias) v += bias[gn];
          if (sc2)  v = v * sc2[gn] + sh2[gn];
          if      (act == ACT_LRELU)   v = lrelu_f(v);
          else if (act == ACT_RELU)    v = fmaxf(v, 0.f);
          else if (act == ACT_GELU)    v = 0.5f * v * (1.f + erff(v * 0.70710678118654752f));
          else if (act == ACT_SIGMOID) v = 1.f / (1.f + expf(-v));
          if (Rb) v += Rb[(long long)gm * ldr + gn];
          Cb[(long long)gm * ldc + gn] = v;
        }
      }
    }
  }
}

// f0 = inpc @ w_in + b_in, written zero-padded to 32 columns (32768 x 32)
__global__ void f0_kernel(const float* __restrict__ inpc, const float* __restrict__ w,
                          const float* __restrict__ bias, float* __restrict__ out)
{
  int t = blockIdx.x * 256 + threadIdx.x;
  if (t >= 32768) return;
  float x = inpc[t * 3], y = inpc[t * 3 + 1], z = inpc[t * 3 + 2];
  float* o = out + (long long)t * 32;
  #pragma unroll
  for (int j = 0; j < 8; ++j)
    o[j] = x * w[j] + y * w[8 + j] + z * w[16 + j] + bias[j];
  #pragma unroll
  for (int j = 8; j < 32; ++j) o[j] = 0.f;
}

// ---------------------------------------------------------------------------
// Brute-force kNN over strided views of inpc (B,2048,3).
// ---------------------------------------------------------------------------
template<int KSEL>
__global__ __launch_bounds__(256) void knn_kernel(const float* __restrict__ pts,
                                                  int Nq, int Nk, int qs, int ks,
                                                  int* __restrict__ out)
{
  int b = blockIdx.y;
  int q = blockIdx.x * 256 + threadIdx.x;
  const float* base = pts + (long long)b * 2048 * 3;
  float qx = 0, qy = 0, qz = 0, qq = 0;
  if (q < Nq) {
    const float* p = base + (long long)q * qs * 3;
    qx = p[0]; qy = p[1]; qz = p[2];
    qq = qx * qx + qy * qy + qz * qz;
  }
  __shared__ float kx[256], ky[256], kz[256], kn[256];
  float bd[KSEL]; int bix[KSEL];
  #pragma unroll
  for (int i = 0; i < KSEL; ++i) { bd[i] = 3.4e38f; bix[i] = 0; }

  for (int j0 = 0; j0 < Nk; j0 += 256) {
    int j = j0 + threadIdx.x;
    if (j < Nk) {
      const float* p = base + (long long)j * ks * 3;
      float x = p[0], y = p[1], z = p[2];
      kx[threadIdx.x] = x; ky[threadIdx.x] = y; kz[threadIdx.x] = z;
      kn[threadIdx.x] = x * x + y * y + z * z;
    }
    __syncthreads();
    int cnt = Nk - j0; if (cnt > 256) cnt = 256;
    if (q < Nq) {
      for (int jj = 0; jj < cnt; ++jj) {
        float d = qq + kn[jj] - 2.f * (qx * kx[jj] + qy * ky[jj] + qz * kz[jj]);
        if (d < bd[KSEL - 1]) {
          bd[KSEL - 1] = d; bix[KSEL - 1] = j0 + jj;
          #pragma unroll
          for (int p = KSEL - 1; p > 0; --p) {
            if (bd[p] < bd[p - 1]) {
              float td = bd[p]; bd[p] = bd[p - 1]; bd[p - 1] = td;
              int   ti = bix[p]; bix[p] = bix[p - 1]; bix[p - 1] = ti;
            }
          }
        }
      }
    }
    __syncthreads();
  }
  if (q < Nq)
    for (int i = 0; i < KSEL; ++i)
      out[((long long)b * Nq + q) * KSEL + i] = bix[i];
}

// Wcat = [W_top | W_bot - W_top] : (2*Din, Dout) -> (Kpad, 2*Dout), zero rows k>=Din
__global__ void wcat_k(const float* __restrict__ W, int Din, int Dout, int Kpad, float* __restrict__ o)
{
  int t = blockIdx.x * 256 + threadIdx.x;
  int total = Kpad * 2 * Dout;
  if (t >= total) return;
  int k = t / (2 * Dout), n = t - k * (2 * Dout);
  float v = 0.f;
  if (k < Din) {
    if (n < Dout) v = W[k * Dout + n];
    else { int nn = n - Dout; v = W[(Din + k) * Dout + nn] - W[k * Dout + nn]; }
  }
  o[t] = v;
}

// out[b,m,d] = max_k lrelu( H[b, idx[b,m,k], d] + H[b, m*qmul, D+d] + bias[d] )
// float4-vectorized over d (D % 4 == 0).
__global__ void edge_maxpool(const float* __restrict__ H, int Nk, int D,
                             const int* __restrict__ idx, int Kn, int qmul,
                             const float* __restrict__ bias, float* __restrict__ out,
                             int ldout, int coff, int Nq, int Bn)
{
  int t = blockIdx.x * 256 + threadIdx.x;
  int D4 = D >> 2;
  int total = Bn * Nq * D4;
  if (t >= total) return;
  int d4 = t % D4; int r = t / D4;
  int m = r % Nq; int b = r / Nq;
  int d = d4 * 4;
  int D2 = 2 * D;
  const float* Hb = H + (long long)b * Nk * D2;
  float4 g2 = *reinterpret_cast<const float4*>(Hb + (long long)(m * qmul) * D2 + D + d);
  float4 bi = *reinterpret_cast<const float4*>(bias + d);
  g2.x += bi.x; g2.y += bi.y; g2.z += bi.z; g2.w += bi.w;
  const int* ip = idx + ((long long)b * Nq + m) * Kn;
  float4 best = make_float4(-3.4e38f, -3.4e38f, -3.4e38f, -3.4e38f);
  for (int kk = 0; kk < Kn; ++kk) {
    float4 v = *reinterpret_cast<const float4*>(Hb + (long long)ip[kk] * D2 + d);
    best.x = fmaxf(best.x, lrelu_f(v.x + g2.x));
    best.y = fmaxf(best.y, lrelu_f(v.y + g2.y));
    best.z = fmaxf(best.z, lrelu_f(v.z + g2.z));
    best.w = fmaxf(best.w, lrelu_f(v.w + g2.w));
  }
  *reinterpret_cast<float4*>(out + ((long long)b * Nq + m) * ldout + coff + d) = best;
}

// layernorm over D=384, one wave per row (wave32 shfl reductions)
__global__ __launch_bounds__(256) void layernorm_k(const float* __restrict__ x,
                                                   const float* __restrict__ g,
                                                   const float* __restrict__ bta,
                                                   float* __restrict__ out, int rows)
{
  int wid = threadIdx.x >> 5, lane = threadIdx.x & 31;
  int row = blockIdx.x * 8 + wid;
  if (row >= rows) return;
  const float* xr = x + (long long)row * 384;
  float vals[12], s = 0.f, sq = 0.f;
  #pragma unroll
  for (int i = 0; i < 12; ++i) { float v = xr[lane + i * 32]; vals[i] = v; s += v; sq += v * v; }
  #pragma unroll
  for (int o = 16; o >= 1; o >>= 1) { s += __shfl_xor(s, o, 32); sq += __shfl_xor(sq, o, 32); }
  float mu = s * (1.f / 384.f);
  float var = sq * (1.f / 384.f) - mu * mu;
  float rs = rsqrtf(var + 1e-5f);
  float* orow = out + (long long)row * 384;
  #pragma unroll
  for (int i = 0; i < 12; ++i) { int d = lane + i * 32; orow[d] = (vals[i] - mu) * rs * g[d] + bta[d]; }
}

// softmax over width 512, one wave per row
__global__ __launch_bounds__(256) void softmax512(float* __restrict__ sc, int rows)
{
  int wid = threadIdx.x >> 5, lane = threadIdx.x & 31;
  int row = blockIdx.x * 8 + wid;
  if (row >= rows) return;
  float* p = sc + (long long)row * 512;
  float vals[16], mx = -3.4e38f;
  #pragma unroll
  for (int i = 0; i < 16; ++i) { float v = p[lane + i * 32]; vals[i] = v; mx = fmaxf(mx, v); }
  #pragma unroll
  for (int o = 16; o >= 1; o >>= 1) mx = fmaxf(mx, __shfl_xor(mx, o, 32));
  float s = 0.f;
  #pragma unroll
  for (int i = 0; i < 16; ++i) { float e = expf(vals[i] - mx); vals[i] = e; s += e; }
  #pragma unroll
  for (int o = 16; o >= 1; o >>= 1) s += __shfl_xor(s, o, 32);
  float inv = 1.f / s;
  #pragma unroll
  for (int i = 0; i < 16; ++i) p[lane + i * 32] = vals[i] * inv;
}

// x += (1-gate)*attn + gate*knn   (attn = gin[:,0:384], knn = gin[:,384:768])
__global__ void combine_k(float* __restrict__ x, const float* __restrict__ gate,
                          const float* __restrict__ gin, int total4)
{
  int t = blockIdx.x * 256 + threadIdx.x;
  if (t >= total4) return;
  int row = t / 96, j = t - row * 96;
  int d = j * 4;
  float4 g  = *reinterpret_cast<const float4*>(gate + (long long)row * 384 + d);
  float4 at = *reinterpret_cast<const float4*>(gin + (long long)row * 768 + d);
  float4 kn = *reinterpret_cast<const float4*>(gin + (long long)row * 768 + 384 + d);
  float4* xp = reinterpret_cast<float4*>(x + (long long)row * 384 + d);
  float4 xv = *xp;
  xv.x += (1.f - g.x) * at.x + g.x * kn.x;
  xv.y += (1.f - g.y) * at.y + g.y * kn.y;
  xv.z += (1.f - g.z) * at.z + g.z * kn.z;
  xv.w += (1.f - g.w) * at.w + g.w * kn.w;
  *xp = xv;
}

__global__ void copy_k(const float* __restrict__ s, float* __restrict__ d, long long n4)
{
  long long t = (long long)blockIdx.x * 256 + threadIdx.x;
  if (t < n4) reinterpret_cast<float4*>(d)[t] = reinterpret_cast<const float4*>(s)[t];
}

// coor2 = inpc[:, ::4, :]  -> (16,512,3)
__global__ void coor_k(const float* __restrict__ inpc, float* __restrict__ o)
{
  int t = blockIdx.x * 256 + threadIdx.x;
  if (t >= 16 * 512 * 3) return;
  int c = t % 3; int m = (t / 3) & 511; int b = t / 1536;
  o[t] = inpc[((long long)b * 2048 + m * 4) * 3 + c];
}

// global_feature = max over 512 rows of y (16,512,1024), float4 over d
__global__ void gmax_k(const float* __restrict__ y, float* __restrict__ gf)
{
  int t = blockIdx.x * 256 + threadIdx.x;
  if (t >= 16 * 256) return;
  int b = t >> 8, d = (t & 255) * 4;
  const float* p = y + (long long)b * 512 * 1024 + d;
  float4 m = *reinterpret_cast<const float4*>(p);
  for (int i = 1; i < 512; ++i) {
    float4 v = *reinterpret_cast<const float4*>(p + (long long)i * 1024);
    m.x = fmaxf(m.x, v.x); m.y = fmaxf(m.y, v.y);
    m.z = fmaxf(m.z, v.z); m.w = fmaxf(m.w, v.w);
  }
  *reinterpret_cast<float4*>(gf + (long long)b * 1024 + d) = m;
}

// ---------------------------------------------------------------------------
// Host side
// ---------------------------------------------------------------------------
static inline unsigned elt(long long n) { return (unsigned)((n + 255) / 256); }

static void gemm_full(hipStream_t st,
    const float* A, long long lda, long long soA, long long siA,
    const float* B, long long ldb, long long soB, long long siB, int transB,
    float* C, long long ldc, long long soC, long long siC,
    int M, int N, int K, int batches, int innerB, float alpha,
    const float* bias, const float* sc2, const float* sh2,
    const float* resid, long long ldr, int act)
{
  dim3 g((N + 127) / 128, (M + 63) / 64, batches);
  wmma_gemm<<<g, 256, 0, st>>>(A, lda, soA, siA, B, ldb, soB, siB, transB,
                               C, ldc, soC, siC, M, N, K, innerB, alpha,
                               bias, sc2, sh2, resid, ldr, act);
}

static void gemm(hipStream_t st, const float* A, int lda, const float* B, int ldb,
                 float* C, int ldc, int M, int N, int K,
                 const float* bias = nullptr, int act = ACT_NONE,
                 const float* resid = nullptr, int ldr = 0,
                 const float* sc2 = nullptr, const float* sh2 = nullptr)
{
  gemm_full(st, A, lda, 0, 0, B, ldb, 0, 0, 0, C, ldc, 0, 0,
            M, N, K, 1, 1, 1.0f, bias, sc2, sh2, resid, ldr, act);
}

extern "C" void kernel_launch(void* const* d_in, const int* in_sizes, int n_in,
                              void* d_out, int out_size, void* d_ws, size_t ws_size,
                              hipStream_t stream)
{
  if (n_in < 46 || !d_out || !d_ws) return;
  const float* inpc  = (const float*)d_in[0];
  const float* w_in  = (const float*)d_in[1];  const float* b_in  = (const float*)d_in[2];
  const float* w1    = (const float*)d_in[3];  const float* b1    = (const float*)d_in[4];
  const float* w2    = (const float*)d_in[5];  const float* b2    = (const float*)d_in[6];
  const float* w3    = (const float*)d_in[7];  const float* b3    = (const float*)d_in[8];
  const float* w4    = (const float*)d_in[9];  const float* b4    = (const float*)d_in[10];
  const float* wp1   = (const float*)d_in[11]; const float* bp1   = (const float*)d_in[12];
  const float* bn_g  = (const float*)d_in[13]; const float* bn_b  = (const float*)d_in[14];
  const float* wp2   = (const float*)d_in[15]; const float* bp2   = (const float*)d_in[16];
  const float* ln1_g = (const float*)d_in[17]; const float* ln1_b = (const float*)d_in[18];
  const float* w_qkv = (const float*)d_in[19];
  const float* w_proj= (const float*)d_in[20]; const float* b_proj= (const float*)d_in[21];
  const float* w_knn = (const float*)d_in[22]; const float* b_knn = (const float*)d_in[23];
  const float* w_g1  = (const float*)d_in[24]; const float* b_g1  = (const float*)d_in[25];
  const float* w_g2  = (const float*)d_in[26]; const float* b_g2  = (const float*)d_in[27];
  const float* ln2_g = (const float*)d_in[28]; const float* ln2_b = (const float*)d_in[29];
  const float* w_m1  = (const float*)d_in[30]; const float* b_m1  = (const float*)d_in[31];
  const float* w_m2  = (const float*)d_in[32]; const float* b_m2  = (const float*)d_in[33];
  const float* r1w1  = (const float*)d_in[34]; const float* r1b1  = (const float*)d_in[35];
  const float* r1w2  = (const float*)d_in[36]; const float* r1b2  = (const float*)d_in[37];
  const float* r1ws  = (const float*)d_in[38]; const float* r1bs  = (const float*)d_in[39];
  const float* r2w1  = (const float*)d_in[40]; const float* r2b1  = (const float*)d_in[41];
  const float* r2w2  = (const float*)d_in[42]; const float* r2b2  = (const float*)d_in[43];
  const float* r2ws  = (const float*)d_in[44]; const float* r2bs  = (const float*)d_in[45];

  int*   wsi = (int*)d_ws;
  float* wsf = (float*)d_ws;

  // Workspace plan (units: 4-byte elements). idx region first, then floats.
  constexpr long long O_IDX1 = 0;                       // 16*2048*16
  constexpr long long O_IDX2 = 524288;                  // 16*1024*16
  constexpr long long O_IDX3 = 786432;
  constexpr long long O_IDX4 = 1048576;                 // 16*512*16
  constexpr long long O_IDX5 = 1179648;                 // 16*512*8
  constexpr long long O_WC1  = 1245184;                 // 32*64 (K-padded)
  constexpr long long O_WC2  = O_WC1 + 2048;            // 32*128
  constexpr long long O_WC3  = O_WC2 + 4096;            // 64*128
  constexpr long long O_WC4  = O_WC3 + 8192;            // 64*256
  constexpr long long O_WCK  = O_WC4 + 16384;           // 384*768
  constexpr long long O_FX   = O_WCK + 294912;          // 1,048,576 (ping; f0 zero-padded 32768x32)
  constexpr long long O_FY   = O_FX + 1048576;          // 1,048,576 (pong)
  constexpr long long O_H    = O_FY + 1048576;          // 4,194,304 (edgeconv GEMM out)
  constexpr long long O_X    = O_H  + 4194304;          // 8192*384
  constexpr long long O_NX   = O_X  + 3145728;          // 8192*384
  constexpr long long O_BIG  = O_NX + 3145728;          // 12,582,912 (qkv / Hk / t2 / t4)
  constexpr long long O_AR   = O_BIG + 12582912;        // 25,165,824 (scores -> g_in/t1/gate -> t3/s1/y1/s2)
  constexpr long long O_ATT  = O_AR + 25165824;         // 8192*384
  constexpr long long TOTALF = O_ATT + 3145728;
  if (ws_size < (size_t)TOTALF * 4) return;

  float* outY   = (float*)d_out;            // (16,512,1024)
  float* outGF  = outY  + 8388608;          // (16,1024)
  float* outCO  = outGF + 16384;            // (16,512,3)
  float* outPRE = outCO + 24576;            // (16,512,384)

  // f0 = inpc @ w_in + b_in, zero-padded to (32768 x 32)
  f0_kernel<<<128, 256, 0, stream>>>(inpc, w_in, b_in, wsf + O_FX);

  // kNN on strided views of inpc
  knn_kernel<16><<<dim3(8, 16), 256, 0, stream>>>(inpc, 2048, 2048, 1, 1, wsi + O_IDX1);
  knn_kernel<16><<<dim3(4, 16), 256, 0, stream>>>(inpc, 1024, 2048, 2, 1, wsi + O_IDX2);
  knn_kernel<16><<<dim3(4, 16), 256, 0, stream>>>(inpc, 1024, 1024, 2, 2, wsi + O_IDX3);
  knn_kernel<16><<<dim3(2, 16), 256, 0, stream>>>(inpc,  512, 1024, 4, 2, wsi + O_IDX4);
  knn_kernel< 8><<<dim3(2, 16), 256, 0, stream>>>(inpc,  512,  512, 4, 4, wsi + O_IDX5);

  // Wcat = [W_top | W_bot - W_top]  (Wcat1 K-padded to 32 rows)
  wcat_k<<<elt(32 * 64),   256, 0, stream>>>(w1,    8,  32, 32, wsf + O_WC1);
  wcat_k<<<elt(32 * 128),  256, 0, stream>>>(w2,   32,  64, 32, wsf + O_WC2);
  wcat_k<<<elt(64 * 128),  256, 0, stream>>>(w3,   64,  64, 64, wsf + O_WC3);
  wcat_k<<<elt(64 * 256),  256, 0, stream>>>(w4,   64, 128, 64, wsf + O_WC4);
  wcat_k<<<elt(384 * 768), 256, 0, stream>>>(w_knn, 384, 384, 384, wsf + O_WCK);

  // edgeconv1: H = f0pad(32768x32) @ Wcat1(32x64) -> gather/max -> f1 (32768 x 32)
  gemm(stream, wsf + O_FX, 32, wsf + O_WC1, 64, wsf + O_H, 64, 32768, 64, 32);
  edge_maxpool<<<elt(16LL * 2048 * 8), 256, 0, stream>>>(wsf + O_H, 2048, 32, wsi + O_IDX1, 16, 1, b1, wsf + O_FY, 32, 0, 2048, 16);
  // edgeconv2 -> f2 (16384 x 64)
  gemm(stream, wsf + O_FY, 32, wsf + O_WC2, 128, wsf + O_H, 128, 32768, 128, 32);
  edge_maxpool<<<elt(16LL * 1024 * 16), 256, 0, stream>>>(wsf + O_H, 2048, 64, wsi + O_IDX2, 16, 2, b2, wsf + O_FX, 64, 0, 1024, 16);
  // edgeconv3 -> f3 (16384 x 64)
  gemm(stream, wsf + O_FX, 64, wsf + O_WC3, 128, wsf + O_H, 128, 16384, 128, 64);
  edge_maxpool<<<elt(16LL * 1024 * 16), 256, 0, stream>>>(wsf + O_H, 1024, 64, wsi + O_IDX3, 16, 1, b3, wsf + O_FY, 64, 0, 1024, 16);
  // edgeconv4 -> f4 (8192 x 128)
  gemm(stream, wsf + O_FY, 64, wsf + O_WC4, 256, wsf + O_H, 256, 16384, 256, 64);
  edge_maxpool<<<elt(16LL * 512 * 32), 256, 0, stream>>>(wsf + O_H, 1024, 128, wsi + O_IDX4, 16, 2, b4, wsf + O_FX, 128, 0, 512, 16);

  // wp1 + BN + lrelu (fused epilogue) -> t ; then wp2 -> x
  gemm(stream, wsf + O_FX, 128, wp1, 384, wsf + O_NX, 384, 8192, 384, 128, bp1, ACT_LRELU, nullptr, 0, bn_g, bn_b);
  gemm(stream, wsf + O_NX, 384, wp2, 384, wsf + O_X, 384, 8192, 384, 384, bp2);

  // ---- salient block ----
  layernorm_k<<<1024, 256, 0, stream>>>(wsf + O_X, ln1_g, ln1_b, wsf + O_NX, 8192);
  gemm(stream, wsf + O_NX, 384, w_qkv, 1152, wsf + O_BIG, 1152, 8192, 1152, 384);

  // scores = alpha * q @ k^T  (96 head-batches, two-level batch strides into qkv)
  gemm_full(stream, wsf + O_BIG, 1152, 589824, 64,
            wsf + O_BIG + 384, 1152, 589824, 64, 1,
            wsf + O_AR, 512, 1572864, 262144,
            512, 512, 64, 96, 6, 0.125f, nullptr, nullptr, nullptr, nullptr, 0, ACT_NONE);
  softmax512<<<6144, 256, 0, stream>>>(wsf + O_AR, 49152);
  // ctx = att @ v  -> attctx (b,m,h*64+d)
  gemm_full(stream, wsf + O_AR, 512, 1572864, 262144,
            wsf + O_BIG + 768, 1152, 589824, 64, 0,
            wsf + O_ATT, 384, 196608, 64,
            512, 64, 512, 96, 6, 1.f, nullptr, nullptr, nullptr, nullptr, 0, ACT_NONE);
  // attn_out -> g_in[:, 0:384]  (scores arena reused as g_in, ldc=768)
  gemm(stream, wsf + O_ATT, 384, w_proj, 384, wsf + O_AR, 768, 8192, 384, 384, b_proj);

  // kNN branch: Hk = nx @ WcatK (8192 x 768), gather/max -> g_in[:, 384:768]
  gemm(stream, wsf + O_NX, 384, wsf + O_WCK, 768, wsf + O_BIG, 768, 8192, 768, 384);
  edge_maxpool<<<elt(16LL * 512 * 96), 256, 0, stream>>>(wsf + O_BIG, 512, 384, wsi + O_IDX5, 8, 1, b_knn, wsf + O_AR, 768, 384, 512, 16);

  // gate = sigmoid(gelu(g_in @ w_g1 + b_g1) @ w_g2 + b_g2)
  const long long O_T1 = O_AR + 6291456, O_GT = O_T1 + 3145728;
  gemm(stream, wsf + O_AR, 768, w_g1, 384, wsf + O_T1, 384, 8192, 384, 768, b_g1, ACT_GELU);
  gemm(stream, wsf + O_T1, 384, w_g2, 384, wsf + O_GT, 384, 8192, 384, 384, b_g2, ACT_SIGMOID);
  combine_k<<<elt(8192LL * 96), 256, 0, stream>>>(wsf + O_X, wsf + O_GT, wsf + O_AR, 8192 * 96);

  // MLP with residual: x += gelu(ln2(x) @ w_m1 + b_m1) @ w_m2 + b_m2
  layernorm_k<<<1024, 256, 0, stream>>>(wsf + O_X, ln2_g, ln2_b, wsf + O_NX, 8192);
  gemm(stream, wsf + O_NX, 384, w_m1, 1536, wsf + O_BIG, 1536, 8192, 1536, 384, b_m1, ACT_GELU);
  gemm(stream, wsf + O_BIG, 1536, w_m2, 384, wsf + O_X, 384, 8192, 384, 1536, b_m2, ACT_NONE, wsf + O_X, 384);

  // outputs: pre and coor
  copy_k<<<elt(786432), 256, 0, stream>>>(wsf + O_X, outPRE, 786432);
  coor_k<<<elt(24576), 256, 0, stream>>>(inpc, outCO);

  // mlp_res1 (384 -> 512)
  const long long O_T3 = O_AR, O_S1 = O_T3 + 4194304, O_Y1 = O_S1 + 4194304, O_S2 = O_Y1 + 4194304;
  gemm(stream, wsf + O_X, 384, r1w1, 512, wsf + O_T3, 512, 8192, 512, 384, r1b1, ACT_RELU);
  gemm(stream, wsf + O_X, 384, r1ws, 512, wsf + O_S1, 512, 8192, 512, 384, r1bs);
  gemm(stream, wsf + O_T3, 512, r1w2, 512, wsf + O_Y1, 512, 8192, 512, 512, r1b2, ACT_NONE, wsf + O_S1, 512);
  // mlp_res2 (512 -> 1024), final GEMM writes y directly into d_out
  gemm(stream, wsf + O_Y1, 512, r2w1, 1024, wsf + O_BIG, 1024, 8192, 1024, 512, r2b1, ACT_RELU);
  gemm(stream, wsf + O_Y1, 512, r2ws, 1024, wsf + O_S2, 1024, 8192, 1024, 512, r2bs);
  gemm(stream, wsf + O_BIG, 1024, r2w2, 1024, outY, 1024, 8192, 1024, 1024, r2b2, ACT_NONE, wsf + O_S2, 1024);

  // global max pool
  gmax_k<<<16, 256, 0, stream>>>(outY, outGF);
}
